// NonLocal2D_74234214744692
// MI455X (gfx1250) — matrix operand order repaired and checked by reading, not compiled, not run on hardware
//
#include <hip/hip_runtime.h>
#include <hip/hip_bf16.h>
#include <stdint.h>

typedef __attribute__((ext_vector_type(16))) _Float16 v16h;
typedef __attribute__((ext_vector_type(8)))  _Float16 v8h;
typedef __attribute__((ext_vector_type(8)))  float    v8f;

#define BATCH 4
#define CCH   256   // C
#define CI    128   // inter channels
#define NPOS  4096  // H*W

// ---------------- WMMA helpers ----------------

__device__ __forceinline__ v8f wmma_f16(v16h a, v16h b, v8f c) {
  return __builtin_amdgcn_wmma_f32_16x16x32_f16(
      /*neg_a=*/false, a, /*neg_b=*/false, b,
      /*c_mod=*/(short)0, c, /*reuse_a=*/false, /*reuse_b=*/false);
}

// A-matrix fragment (16 rows x 32 K, 16-bit), row-major source with leading dim ld (halves).
// Per ISA: lanes 0-15 -> row M=lane, halves {K0..7, K16..23}; lanes 16-31 -> row M=lane-16,
// halves {K8..15, K24..31}.
__device__ __forceinline__ v16h load_frag_a(const _Float16* __restrict__ base, int ld, int lane) {
  int r  = lane & 15;
  int kb = (lane >> 4) << 3;            // 0 or 8
  const v8h* p0 = (const v8h*)(base + r * ld + kb);
  const v8h* p1 = (const v8h*)(base + r * ld + kb + 16);
  v8h a = *p0, b = *p1;
  v16h o;
#pragma unroll
  for (int i = 0; i < 8; ++i) { o[i] = a[i]; o[i + 8] = b[i]; }
  return o;
}

// B-matrix fragment (32 K x 16 cols, 16-bit). Source stored column-major over K:
// row index = output column n, contiguous K. Lanes 0-15 hold K{0..7,16..23} of column n=lane,
// lanes 16-31 hold K{8..15,24..31}.
__device__ __forceinline__ v16h load_frag_b(const _Float16* __restrict__ base, int ld, int lane) {
  int r  = lane & 15;
  int kb = (lane >> 4) << 3;            // 0 or 8
  const v8h* p0 = (const v8h*)(base + r * ld + kb);
  const v8h* p1 = (const v8h*)(base + r * ld + kb + 16);
  v8h a = *p0, b = *p1;
  v16h o;
#pragma unroll
  for (int i = 0; i < 8; ++i) { o[i] = a[i]; o[i + 8] = b[i]; }
  return o;
}

// ---------------- Tensor Data Mover: 2D tile (rows x 128 f16) -> padded LDS ----------------
// Copies `rows` rows of 256B (64 dwords) each, row stride 256B in memory, inserting a 16B
// LDS pad after every 256B stored -> LDS leading dim of 136 halves. D# built per ISA 8.3/8.4.

__device__ __forceinline__ void tdm_load_tile(uint32_t lds_off, uint64_t ga, int rows) {
  typedef __attribute__((ext_vector_type(4))) unsigned int u32x4;
  typedef __attribute__((ext_vector_type(8))) int i32x8;
  typedef __attribute__((ext_vector_type(4))) int i32x4;
  u32x4 g0;
  g0[0] = 1u;                                                  // count=1 (valid), user mode
  g0[1] = lds_off;                                             // lds_addr (bytes)
  g0[2] = (uint32_t)ga;                                        // global_addr[31:0]
  g0[3] = (uint32_t)((ga >> 32) & 0x01FFFFFFu) | (2u << 30);   // global_addr[56:32] | type=2
  i32x8 g1;
  g1[0] = (int)((2u << 16)                                     // data_size = 4B
              | (1u << 20)                                     // pad_enable
              | (5u << 22)                                     // pad_interval: every 256B
              | (3u << 25));                                   // pad_amount: 4 dwords (16B)
  g1[1] = (int)(64u << 16);                                    // tensor_dim0 = 64 dwords
  g1[2] = (int)((4096u & 0xFFFFu) << 16);                      // tensor_dim1 lo = 4096
  g1[3] = (int)(64u << 16);                                    // tile_dim0 = 64 dwords
  g1[4] = (int)(rows & 0xFFFF);                                // tile_dim1 = rows
  g1[5] = 64;                                                  // tensor_dim0_stride = 64 dw
  g1[6] = 0;
  g1[7] = 0;
  i32x4 z4 = {0, 0, 0, 0};
#if __clang_major__ >= 23
  i32x8 z8 = {0, 0, 0, 0, 0, 0, 0, 0};
  __builtin_amdgcn_tensor_load_to_lds(g0, g1, z4, z4, z8, 0);
#else
  __builtin_amdgcn_tensor_load_to_lds(g0, g1, z4, z4, 0);
#endif
}

// ---------------- Kernel 0a: spatial mean of x per (b,c) ----------------

__global__ __launch_bounds__(256) void k_xbar(const float* __restrict__ x,
                                              float* __restrict__ xbar) {
  __shared__ float red[256];
  int bc = blockIdx.x;                               // b*C + c, 1024 blocks
  const float* p = x + (size_t)bc * NPOS;
  float s = 0.f;
  for (int i = threadIdx.x; i < NPOS; i += 256) s += p[i];
  red[threadIdx.x] = s;
  __syncthreads();
  for (int w = 128; w > 0; w >>= 1) {
    if ((int)threadIdx.x < w) red[threadIdx.x] += red[threadIdx.x + w];
    __syncthreads();
  }
  if (threadIdx.x == 0) xbar[bc] = red[0] * (1.f / NPOS);
}

// ---------------- Kernel 0b: weight prep (f16 convert, centering, phibar) ----------------

__global__ __launch_bounds__(256) void k_prep(
    const float* __restrict__ g_w,  const float* __restrict__ theta_w,
    const float* __restrict__ theta_b, const float* __restrict__ phi_w,
    const float* __restrict__ out_w, const float* __restrict__ xbar,
    _Float16* __restrict__ gw16, _Float16* __restrict__ thw16,
    _Float16* __restrict__ phw16, _Float16* __restrict__ outw16,
    float* __restrict__ tb_c, float* __restrict__ phibar) {
  int tid = threadIdx.x;
  __shared__ float colmean[CCH];
  {
    float s = 0.f;
    for (int o = 0; o < CI; ++o) s += theta_w[o * CCH + tid];
    colmean[tid] = s * (1.f / CI);
  }
  __syncthreads();
  for (int i = tid; i < CI * CCH; i += 256) {
    gw16[i]  = (_Float16)g_w[i];
    phw16[i] = (_Float16)phi_w[i];
    thw16[i] = (_Float16)(theta_w[i] - colmean[i & (CCH - 1)]);
  }
  for (int i = tid; i < CCH * CI; i += 256) outw16[i] = (_Float16)out_w[i];
  if (tid < CI) {
    float m = 0.f;
    for (int o = 0; o < CI; ++o) m += theta_b[o];
    m *= (1.f / CI);
    tb_c[tid] = theta_b[tid] - m;
  }
  for (int i = tid; i < BATCH * CI; i += 256) {
    int b = i >> 7, o = i & (CI - 1);
    float s = 0.f;
    for (int c = 0; c < CCH; ++c) s += phi_w[o * CCH + c] * xbar[b * CCH + c];
    phibar[i] = s;
  }
}

// ---------------- Kernel 1: projection GEMMs -> T, P(=phi^T), G as (N x CI) f16 ----------------

__global__ __launch_bounds__(256) void k_proj(
    const float* __restrict__ x,
    const _Float16* __restrict__ gw16, const _Float16* __restrict__ thw16,
    const _Float16* __restrict__ phw16,
    const float* __restrict__ g_b, const float* __restrict__ tb_c,
    const float* __restrict__ phibar,
    _Float16* __restrict__ T, _Float16* __restrict__ P, _Float16* __restrict__ G) {
  __shared__ _Float16 xt[32 * 264];                  // x^T tile: [n][c], ld=264
  int b     = blockIdx.x >> 7;                       // B * 128 blocks
  int nbase = (blockIdx.x & 127) * 32;
  int tid = threadIdx.x, wave = tid >> 5, lane = tid & 31;
  const float* xb = x + (size_t)b * CCH * NPOS;

  {
    int n = tid & 31;
    for (int c = tid >> 5; c < CCH; c += 8)
      xt[n * 264 + c] = (_Float16)xb[(size_t)c * NPOS + nbase + n];
  }
  __syncthreads();

  for (int t = wave; t < 48; t += 8) {               // 3 projections x 2 n-tiles x 8 o-tiles
    int proj = t >> 4;
    int rem  = t & 15;
    int nt   = rem >> 3, ot = rem & 7;
    const _Float16* W = (proj == 0) ? gw16 : (proj == 1) ? thw16 : phw16;
    v8f acc = {};
#pragma unroll
    for (int k = 0; k < 8; ++k) {                    // K = C = 256
      v16h a  = load_frag_a(&xt[(nt * 16) * 264 + k * 32], 264, lane);
      v16h bf = load_frag_b(W + (ot * 16) * CCH + k * 32, CCH, lane);
      acc = wmma_f16(a, bf, acc);
    }
    int o = ot * 16 + (lane & 15);
    float bias = (proj == 0) ? g_b[o] : (proj == 1) ? tb_c[o] : -phibar[b * CI + o];
    _Float16* D = ((proj == 0) ? G : (proj == 1) ? T : P) + (size_t)b * NPOS * CI;
    int Mb = (lane >> 4) * 8;
#pragma unroll
    for (int r = 0; r < 8; ++r) {
      int n = nbase + nt * 16 + r + Mb;
      D[(size_t)n * CI + o] = (_Float16)(acc[r] + bias);
    }
  }
}

// ---------------- Kernel 2: fused pairwise @ g, out-projection, residual ----------------

__global__ __launch_bounds__(256) void k_main(
    const _Float16* __restrict__ T, const _Float16* __restrict__ P,
    const _Float16* __restrict__ G, const _Float16* __restrict__ outw16,
    const float* __restrict__ out_b, const float* __restrict__ x,
    float* __restrict__ out) {
  __shared__ _Float16 ph[32 * 136];                  // phi^T tile [m][ci], ld=136 (TDM-filled)
  __shared__ _Float16 gt[128 * 40];                  // g tile transposed [ci][m], ld=40
  __shared__ _Float16 pw[8][16 * 32];                // per-wave pairwise scratch [n][m]
  __shared__ _Float16 ylds[128 * 136];               // y tile [n][ci], ld=136

  int b     = blockIdx.x >> 5;                       // 4 * 32 blocks
  int nbase = (blockIdx.x & 31) * 128;               // 128 n rows per WG
  int tid = threadIdx.x, wave = tid >> 5, lane = tid & 31;

  const _Float16* Tb = T + (size_t)b * NPOS * CI;
  const _Float16* Pb = P + (size_t)b * NPOS * CI;
  const _Float16* Gb = G + (size_t)b * NPOS * CI;

  uint32_t ph_lds_off =
      (uint32_t)(uintptr_t)(__attribute__((address_space(3))) _Float16*)&ph[0];

  // Persistent theta A fragments for this wave's 16 n-rows (K = CI = 128)
  v16h ta[4];
#pragma unroll
  for (int k = 0; k < 4; ++k)
    ta[k] = load_frag_a(Tb + (size_t)(nbase + wave * 16) * CI + k * 32, CI, lane);

  v8f yacc[8];
#pragma unroll
  for (int i = 0; i < 8; ++i) { v8f z = {}; yacc[i] = z; }

  int lrow = tid >> 3;                               // 0..31
  int lcol = (tid & 7) << 4;                         // 0..112

  for (int mb = 0; mb < NPOS; mb += 32) {
    __syncthreads();
    // async TDM: phi^T tile (32 x 128 f16, padded ld=136) memory -> LDS
    if (tid == 0)
      tdm_load_tile(ph_lds_off, (uint64_t)(uintptr_t)(Pb + (size_t)mb * CI), 32);
    {  // overlap: stage transposed g tile manually
      const v8h* gs = (const v8h*)(Gb + (size_t)(mb + lrow) * CI + lcol);
      v8h g0 = gs[0], g1 = gs[1];
#pragma unroll
      for (int i = 0; i < 8; ++i) {
        gt[(lcol + i) * 40 + lrow]     = g0[i];
        gt[(lcol + 8 + i) * 40 + lrow] = g1[i];
      }
      if (mb + 32 < NPOS)                            // global_prefetch_b8 for next g tile
        __builtin_prefetch(Gb + (size_t)(mb + 32 + lrow) * CI + lcol, 0, 1);
    }
    __builtin_amdgcn_s_wait_tensorcnt(0);
    __syncthreads();

    // pairwise tile: 16 n x 32 m  (two 16x16 halves), K = CI via 4 WMMAs each
#pragma unroll
    for (int half = 0; half < 2; ++half) {
      v8f acc = {};
#pragma unroll
      for (int k = 0; k < 4; ++k) {
        v16h bf = load_frag_b(&ph[(half * 16) * 136 + k * 32], 136, lane);
        acc = wmma_f16(ta[k], bf, acc);
      }
      int Nc = lane & 15;
      int Mb = (lane >> 4) * 8;
#pragma unroll
      for (int r = 0; r < 8; ++r)
        pw[wave][(r + Mb) * 32 + half * 16 + Nc] = (_Float16)acc[r];
    }
    // per-wave LDS round-trip: make cross-lane pairwise data visible
    asm volatile("s_wait_dscnt 0" ::: "memory");

    v16h pa = load_frag_a(pw[wave], 32, lane);       // A = pairwise (16 n x 32 m)
#pragma unroll
    for (int ci = 0; ci < 8; ++ci) {                 // y[n, ci-tile] += pa @ g
      v16h gb = load_frag_b(&gt[(ci * 16) * 40], 40, lane);
      yacc[ci] = wmma_f16(pa, gb, yacc[ci]);
    }
  }

  // spill y (f32 acc -> f16) into LDS for the out-projection
  {
    int Nc = lane & 15;
    int Mb = (lane >> 4) * 8;
#pragma unroll
    for (int ci = 0; ci < 8; ++ci)
#pragma unroll
      for (int r = 0; r < 8; ++r)
        ylds[(wave * 16 + r + Mb) * 136 + ci * 16 + Nc] = (_Float16)yacc[ci][r];
  }
  __syncthreads();

  // out[b, o, n] = out_w @ y^T + out_b + x ; wave handles its own 16 n-cols, all 256 o
  const float* xb = x + (size_t)b * CCH * NPOS;
  float* ob = out + (size_t)b * CCH * NPOS;
  int ncol = nbase + wave * 16 + (lane & 15);
#pragma unroll 1
  for (int ot = 0; ot < 16; ++ot) {
    v8f acc = {};
#pragma unroll
    for (int k = 0; k < 4; ++k) {
      v16h wa = load_frag_a(outw16 + (ot * 16) * CI + k * 32, CI, lane);
      v16h yb = load_frag_b(&ylds[(wave * 16) * 136 + k * 32], 136, lane);
      acc = wmma_f16(wa, yb, acc);
    }
    int Mb = (lane >> 4) * 8;
#pragma unroll
    for (int r = 0; r < 8; ++r) {
      int o = ot * 16 + r + Mb;
      ob[(size_t)o * NPOS + ncol] = acc[r] + out_b[o] + xb[(size_t)o * NPOS + ncol];
    }
  }
}

// ---------------- launch ----------------

extern "C" void kernel_launch(void* const* d_in, const int* in_sizes, int n_in,
                              void* d_out, int out_size, void* d_ws, size_t ws_size,
                              hipStream_t stream) {
  const float* x       = (const float*)d_in[0];
  const float* g_w     = (const float*)d_in[1];
  const float* g_b     = (const float*)d_in[2];
  const float* theta_w = (const float*)d_in[3];
  const float* theta_b = (const float*)d_in[4];
  const float* phi_w   = (const float*)d_in[5];
  /* phi_b (d_in[6]) cancels under spatial centering */
  const float* out_w   = (const float*)d_in[7];
  const float* out_b   = (const float*)d_in[8];
  float* out = (float*)d_out;

  uint8_t* ws = (uint8_t*)d_ws;
  size_t off = 0;
  const size_t SZP = (size_t)BATCH * NPOS * CI * sizeof(_Float16);  // 4 MB each
  _Float16* T      = (_Float16*)(ws + off); off += SZP;
  _Float16* P      = (_Float16*)(ws + off); off += SZP;
  _Float16* G      = (_Float16*)(ws + off); off += SZP;
  _Float16* gw16   = (_Float16*)(ws + off); off += (size_t)CI * CCH * 2;
  _Float16* thw16  = (_Float16*)(ws + off); off += (size_t)CI * CCH * 2;
  _Float16* phw16  = (_Float16*)(ws + off); off += (size_t)CI * CCH * 2;
  _Float16* outw16 = (_Float16*)(ws + off); off += (size_t)CCH * CI * 2;
  float* tb_c   = (float*)(ws + off); off += 1024;
  float* xbar   = (float*)(ws + off); off += (size_t)BATCH * CCH * 4;
  float* phibar = (float*)(ws + off); off += (size_t)BATCH * CI * 4;
  (void)in_sizes; (void)n_in; (void)out_size; (void)ws_size;

  k_xbar<<<BATCH * CCH, 256, 0, stream>>>(x, xbar);
  k_prep<<<1, 256, 0, stream>>>(g_w, theta_w, theta_b, phi_w, out_w, xbar,
                                gw16, thw16, phw16, outw16, tb_c, phibar);
  k_proj<<<BATCH * (NPOS / 32), 256, 0, stream>>>(x, gw16, thw16, phw16,
                                                  g_b, tb_c, phibar, T, P, G);
  k_main<<<BATCH * (NPOS / 128), 256, 0, stream>>>(T, P, G, outw16, out_b, x, out);
}
